// Net_40965398069469
// MI455X (gfx1250) — matrix operand order repaired and checked by reading, not compiled, not run on hardware
//
#include <hip/hip_runtime.h>
#include <math.h>

typedef __bf16 bf16_t;
typedef __attribute__((ext_vector_type(16))) __bf16 v16bf;
typedef __attribute__((ext_vector_type(8)))  __bf16 v8bf;
typedef __attribute__((ext_vector_type(8)))  float  v8f;

#define ENC_NEGINF 0x007FFFFFu   // encode(-inf)

// ---------- helpers ----------
__device__ __forceinline__ unsigned enc_f32(float f) {
  unsigned b = __float_as_uint(f);
  return (b & 0x80000000u) ? ~b : (b | 0x80000000u);
}
__device__ __forceinline__ float dec_f32(unsigned u) {
  unsigned b = (u & 0x80000000u) ? (u & 0x7FFFFFFFu) : ~u;
  return __uint_as_float(b);
}
__device__ __forceinline__ float elu1(float v) { return v > 0.f ? v : expm1f(v); }

// spline basis for one tap s (0..7), ks=5, 3 pseudo dims
__device__ __forceinline__ void spline_tap(const float* ps, int s, float* basis, int* kidx) {
  float b = 1.f; int idx = 0; int stride = 1;
#pragma unroll
  for (int d = 0; d < 3; ++d) {
    float p  = ps[d] * 4.0f;
    float lo = floorf(p);
    lo = fminf(fmaxf(lo, 0.f), 3.f);
    float f  = p - lo;
    int bit  = (s >> d) & 1;
    b *= bit ? f : (1.f - f);
    idx += ((int)lo + bit) * stride;
    stride *= 5;
  }
  *basis = b; *kidx = idx;
}

// ---------- generic small kernels ----------
__global__ void fill_u32(unsigned* p, unsigned v, int n) {
  int i = blockIdx.x * blockDim.x + threadIdx.x;
  if (i < n) p[i] = v;
}

// Convert B (batches x K x N, f32 row-major) into bf16 WMMA-fragment order:
// out[(((b*NT + ct)*KC + kc)*32 + L)*16 + e] = B[b][kc*32 + e + (L>=16)*16][ct*16 + (L&15)]
// so each lane's 16-element B fragment is contiguous (32B -> 2x b128 loads).
__global__ void cvt_swizzleB(const float* B, bf16_t* out, int K, int N, int batches) {
  long i = (long)blockIdx.x * blockDim.x + threadIdx.x;
  long total = (long)batches * K * N;
  if (i >= total) return;
  int  e  = (int)(i & 15);
  int  L  = (int)((i >> 4) & 31);
  long r  = i >> 9;
  int  KC = K >> 5, NT = N >> 4;
  int  kc = (int)(r % KC); r /= KC;
  int  ct = (int)(r % NT);
  long b  = r / NT;
  int  kk = kc * 32 + e + ((L >= 16) ? 16 : 0);
  int  nn = ct * 16 + (L & 15);
  out[i] = (bf16_t)B[(b * K + kk) * (long)N + nn];
}

// segment max: thread per (node, channel), monotone-encoded u32 atomicMax
__global__ void segmax_atomic(const float* x, const int* cl, unsigned* enc, int n, int C) {
  int i = blockIdx.x * blockDim.x + threadIdx.x;
  if (i >= n * C) return;
  int node = i / C, c = i - node * C;
  atomicMax(&enc[(long)cl[node] * C + c], enc_f32(x[i]));
}

__global__ void segmax_decode(const unsigned* enc, float* out_f, bf16_t* out_b, int n) {
  int i = blockIdx.x * blockDim.x + threadIdx.x;
  if (i >= n) return;
  float v = dec_f32(enc[i]);
  if (!isfinite(v)) v = 0.f;            // empty segments -> 0 (matches reference)
  out_f[i] = v;
  if (out_b) out_b[i] = (bf16_t)v;
}

// ---------- WMMA bf16 GEMM core (wave32, one 16x16 tile per 32-thread block) ----------
// A: row-major M x K (lda == K, 16B-aligned rows). Bfrag: pre-swizzled fragment stream.
template<int K>
__device__ __forceinline__ v8f wmma_tile_sw(const bf16_t* A, const bf16_t* Bfrag, int row0) {
  constexpr int KC = K / 32;
  v8f c = {};
  int L    = threadIdx.x & 31;
  int half = (L >= 16) ? 1 : 0;
  int m    = row0 + (L & 15);
  const bf16_t* arow = A + (long)m * K + half * 16;
  const bf16_t* bptr = Bfrag + L * 16;
#pragma unroll
  for (int kc = 0; kc < KC; ++kc) {
    v8bf lo = *(const v8bf*)(arow + kc * 32);        // K = kc*32 + half*16 + [0..7]
    v8bf hi = *(const v8bf*)(arow + kc * 32 + 8);    // K = kc*32 + half*16 + [8..15]
    v16bf a = __builtin_shufflevector(lo, hi, 0, 1, 2, 3, 4, 5, 6, 7,
                                              8, 9, 10, 11, 12, 13, 14, 15);
    v16bf b = *(const v16bf*)(bptr + (long)kc * 512);
    c = __builtin_amdgcn_wmma_f32_16x16x32_bf16(false, a, false, b, (short)0, c, false, false);
  }
  return c;
}

// batched over blockIdx.z (125 spline kernels): bf16 output at C[row*ldc + z*cstride + col]
template<int K>
__global__ void gemm_sw_bf16(const bf16_t* A, const bf16_t* Bsw, int NT,
                             bf16_t* Cc, int ldc, long cstride) {
  constexpr int KC = K / 32;
  const bf16_t* Bfrag = Bsw + ((long)blockIdx.z * NT + blockIdx.y) * (KC * 512);
  bf16_t*       Cp    = Cc + (long)blockIdx.z * cstride;
  int row0 = blockIdx.x * 16, col0 = blockIdx.y * 16;
  v8f c = wmma_tile_sw<K>(A, Bfrag, row0);
  int L = threadIdx.x & 31;
  int half = (L >= 16) ? 1 : 0;
  int n = col0 + (L & 15);
#pragma unroll
  for (int r = 0; r < 8; ++r)
    Cp[(long)(row0 + r + half * 8) * ldc + n] = (bf16_t)c[r];
}

// f32 output variant with optional bias + ELU (root GEMMs / FC1)
template<int K>
__global__ void gemm_sw_f32(const bf16_t* A, const bf16_t* Bsw,
                            float* Cc, int ldc, const float* bias, int act) {
  constexpr int KC = K / 32;
  const bf16_t* Bfrag = Bsw + (long)blockIdx.y * (KC * 512);
  int row0 = blockIdx.x * 16, col0 = blockIdx.y * 16;
  v8f c = wmma_tile_sw<K>(A, Bfrag, row0);
  int L = threadIdx.x & 31;
  int half = (L >= 16) ? 1 : 0;
  int n = col0 + (L & 15);
  float bv = bias ? bias[n] : 0.f;
#pragma unroll
  for (int r = 0; r < 8; ++r) {
    float v = c[r] + bv;
    if (act) v = elu1(v);
    Cc[(long)(row0 + r + half * 8) * ldc + n] = v;
  }
}

// ---------- spline conv layer 1 (in_channels = 1) ----------
__global__ void spline_conv1(const float* xin, const int* src, const int* dst,
                             const float* pseudo, const float* W1, // 125x64
                             float* agg, float* deg, int E) {
  __shared__ float sb[4][8];
  __shared__ int   sk[4][8];
  int g = threadIdx.x >> 6;        // 4 edges per 256-thread block
  int lane = threadIdx.x & 63;     // channel
  int e = blockIdx.x * 4 + g;
  if (e < E && lane < 8) spline_tap(&pseudo[(long)e * 3], lane, &sb[g][lane], &sk[g][lane]);
  __syncthreads();
  if (e >= E) return;
  float ws = 0.f;
#pragma unroll
  for (int t = 0; t < 8; ++t) ws += sb[g][t] * W1[sk[g][t] * 64 + lane];
  float val = xin[src[e]] * ws;
  atomicAdd(&agg[(long)dst[e] * 64 + lane], val);
  if (lane == 0) atomicAdd(&deg[dst[e]], 1.0f);
}

__global__ void finalize1(const float* agg, const float* deg, const float* xin,
                          const float* root1, const float* b1, float* xout, int n) {
  int i = blockIdx.x * blockDim.x + threadIdx.x;
  if (i >= n * 64) return;
  int node = i >> 6, o = i & 63;
  float v = agg[i] / fmaxf(deg[node], 1.f) + xin[node] * root1[o] + b1[o];
  xout[i] = elu1(v);
}

// ---------- spline conv layers 2/3: gather 8 taps from bf16 XW, scatter fp32 ----------
__global__ void spline_conv_edges(const bf16_t* XW, const int* src, const int* dst,
                                  const float* pseudo, float* agg, float* deg,
                                  int E, int C, int kstride) {
  __shared__ float sb[4][8];
  __shared__ int   sk[4][8];
  int groups = blockDim.x / C;     // 4 (C=64) or 2 (C=128)
  int g = threadIdx.x / C;
  int lane = threadIdx.x - g * C;
  int e = blockIdx.x * groups + g;
  if (e < E && lane < 8) spline_tap(&pseudo[(long)e * 3], lane, &sb[g][lane], &sk[g][lane]);
  __syncthreads();
  if (e >= E) return;
  const bf16_t* bse = XW + (long)src[e] * kstride;
  float acc = 0.f;
#pragma unroll
  for (int t = 0; t < 8; ++t) acc += sb[g][t] * (float)bse[sk[g][t] * C + lane];
  atomicAdd(&agg[(long)dst[e] * C + lane], acc);
  if (lane == 0) atomicAdd(&deg[dst[e]], 1.0f);
}

__global__ void finalize_g(const float* agg, const float* deg, const float* rootout,
                           const float* bias, float* xout, int n, int C) {
  int i = blockIdx.x * blockDim.x + threadIdx.x;
  if (i >= n * C) return;
  int node = i / C, o = i - node * C;
  float v = agg[i] / fmaxf(deg[node], 1.f) + rootout[i] + bias[o];
  xout[i] = elu1(v);
}

// ---------- FC2 + log_softmax, one block per batch row ----------
__global__ void fc2_logsoftmax(const float* h, const float* w, const float* b, float* out) {
  __shared__ float z[10];
  __shared__ float lse;
  int bi = blockIdx.x, t = threadIdx.x;
  if (t < 10) {
    float acc = b[t];
    for (int k = 0; k < 256; ++k) acc += h[bi * 256 + k] * w[k * 10 + t];
    z[t] = acc;
  }
  __syncthreads();
  if (t == 0) {
    float m = z[0];
    for (int i = 1; i < 10; ++i) m = fmaxf(m, z[i]);
    float s = 0.f;
    for (int i = 0; i < 10; ++i) s += expf(z[i] - m);
    lse = m + logf(s);
  }
  __syncthreads();
  if (t < 10) out[bi * 10 + t] = z[t] - lse;
}

// ---------- host driver ----------
extern "C" void kernel_launch(void* const* d_in, const int* in_sizes, int n_in,
                              void* d_out, int out_size, void* d_ws, size_t ws_size,
                              hipStream_t stream) {
  (void)in_sizes; (void)n_in; (void)out_size; (void)ws_size;
  const float* x0    = (const float*)d_in[0];
  const int*   cl0   = (const int*)d_in[1];
  const int*   e1p   = (const int*)d_in[2];
  const float* ps1   = (const float*)d_in[3];
  const float* W1    = (const float*)d_in[4];
  const float* root1 = (const float*)d_in[5];
  const float* b1    = (const float*)d_in[6];
  const int*   cl1   = (const int*)d_in[7];
  const int*   e2p   = (const int*)d_in[8];
  const float* ps2   = (const float*)d_in[9];
  const float* W2    = (const float*)d_in[10];
  const float* root2 = (const float*)d_in[11];
  const float* b2    = (const float*)d_in[12];
  const int*   cl2   = (const int*)d_in[13];
  const int*   e3p   = (const int*)d_in[14];
  const float* ps3   = (const float*)d_in[15];
  const float* W3    = (const float*)d_in[16];
  const float* root3 = (const float*)d_in[17];
  const float* b3    = (const float*)d_in[18];
  const int*   cl3   = (const int*)d_in[19];
  const float* fc1w  = (const float*)d_in[20];
  const float* fc1b  = (const float*)d_in[21];
  const float* fc2w  = (const float*)d_in[22];
  const float* fc2b  = (const float*)d_in[23];
  float* out = (float*)d_out;

  const int N0 = 120000, N1 = 20000, N2 = 6000, N3 = 1600;
  const int E1 = 480000, E2 = 144000, E3 = 38400;

  char* base = (char*)d_ws;
  size_t off = 0;
  auto alloc = [&](size_t bytes) -> void* {
    void* p = base + off;
    off += (bytes + 255) & ~(size_t)255;
    return p;
  };
  auto cdiv = [](int a, int b) { return (a + b - 1) / b; };

  float*    x1in = (float*)alloc((size_t)N1 * 4);
  unsigned* enc  = (unsigned*)alloc((size_t)N2 * 64 * 4);            // biggest pool target (384000)
  float*    x1   = (float*)alloc((size_t)N1 * 64 * 4);
  float*    agg1 = (float*)alloc((size_t)N1 * 64 * 4);
  float*    deg1 = (float*)alloc((size_t)N1 * 4);
  float*    x2in = (float*)alloc((size_t)N2 * 64 * 4);
  bf16_t*   x2bf = (bf16_t*)alloc((size_t)N2 * 64 * 2);
  bf16_t*   W2bf = (bf16_t*)alloc((size_t)125 * 64 * 64 * 2);        // swizzled fragments
  bf16_t*   r2bf = (bf16_t*)alloc((size_t)64 * 64 * 2);
  bf16_t*   XWb  = (bf16_t*)alloc((size_t)48000000 * 2);             // max(N2*125*64, N3*125*128), L2-resident
  float*    agg2 = (float*)alloc((size_t)N2 * 64 * 4);
  float*    deg2 = (float*)alloc((size_t)N2 * 4);
  float*    ro2  = (float*)alloc((size_t)N2 * 64 * 4);
  float*    x2o  = (float*)alloc((size_t)N2 * 64 * 4);
  float*    x3in = (float*)alloc((size_t)N3 * 64 * 4);
  bf16_t*   x3bf = (bf16_t*)alloc((size_t)N3 * 64 * 2);
  bf16_t*   W3bf = (bf16_t*)alloc((size_t)125 * 64 * 128 * 2);
  bf16_t*   r3bf = (bf16_t*)alloc((size_t)64 * 128 * 2);
  float*    agg3 = (float*)alloc((size_t)N3 * 128 * 4);
  float*    deg3 = (float*)alloc((size_t)N3 * 4);
  float*    ro3  = (float*)alloc((size_t)N3 * 128 * 4);
  float*    x3o  = (float*)alloc((size_t)N3 * 128 * 4);
  float*    xfc  = (float*)alloc((size_t)512 * 128 * 4);
  bf16_t*   xfcbf= (bf16_t*)alloc((size_t)512 * 128 * 2);
  bf16_t*   fc1bf= (bf16_t*)alloc((size_t)1024 * 256 * 2);
  float*    h1   = (float*)alloc((size_t)64 * 256 * 4);

  // ---- pool0: x0 (120000,1) -> x1in (20000,1)
  fill_u32<<<cdiv(N1, 256), 256, 0, stream>>>(enc, ENC_NEGINF, N1);
  segmax_atomic<<<cdiv(N0, 256), 256, 0, stream>>>(x0, cl0, enc, N0, 1);
  segmax_decode<<<cdiv(N1, 256), 256, 0, stream>>>(enc, x1in, nullptr, N1);

  // ---- layer 1 (in=1, out=64): edge scatter + finalize
  fill_u32<<<cdiv(N1 * 64, 256), 256, 0, stream>>>((unsigned*)agg1, 0u, N1 * 64);
  fill_u32<<<cdiv(N1, 256), 256, 0, stream>>>((unsigned*)deg1, 0u, N1);
  spline_conv1<<<cdiv(E1, 4), 256, 0, stream>>>(x1in, e1p, e1p + E1, ps1, W1, agg1, deg1, E1);
  finalize1<<<cdiv(N1 * 64, 256), 256, 0, stream>>>(agg1, deg1, x1in, root1, b1, x1, N1);

  // ---- pool1 -> x2in (6000,64) f32 + bf16
  fill_u32<<<cdiv(N2 * 64, 256), 256, 0, stream>>>(enc, ENC_NEGINF, N2 * 64);
  segmax_atomic<<<cdiv(N1 * 64, 256), 256, 0, stream>>>(x1, cl1, enc, N1, 64);
  segmax_decode<<<cdiv(N2 * 64, 256), 256, 0, stream>>>(enc, x2in, x2bf, N2 * 64);

  // ---- layer 2: 125x batched WMMA GEMM -> bf16 XW (96MB, fits L2), root GEMM, gather/scatter
  cvt_swizzleB<<<cdiv(125 * 64 * 64, 256), 256, 0, stream>>>(W2, W2bf, 64, 64, 125);
  cvt_swizzleB<<<cdiv(64 * 64, 256), 256, 0, stream>>>(root2, r2bf, 64, 64, 1);
  gemm_sw_bf16<64><<<dim3(N2 / 16, 4, 125), 32, 0, stream>>>(x2bf, W2bf, 4,
                                                             XWb, 125 * 64, (long)64);
  gemm_sw_f32<64><<<dim3(N2 / 16, 4), 32, 0, stream>>>(x2bf, r2bf, ro2, 64, nullptr, 0);
  fill_u32<<<cdiv(N2 * 64, 256), 256, 0, stream>>>((unsigned*)agg2, 0u, N2 * 64);
  fill_u32<<<cdiv(N2, 256), 256, 0, stream>>>((unsigned*)deg2, 0u, N2);
  spline_conv_edges<<<cdiv(E2, 4), 256, 0, stream>>>(XWb, e2p, e2p + E2, ps2, agg2, deg2,
                                                     E2, 64, 125 * 64);
  finalize_g<<<cdiv(N2 * 64, 256), 256, 0, stream>>>(agg2, deg2, ro2, b2, x2o, N2, 64);

  // ---- pool2 -> x3in (1600,64)
  fill_u32<<<cdiv(N3 * 64, 256), 256, 0, stream>>>(enc, ENC_NEGINF, N3 * 64);
  segmax_atomic<<<cdiv(N2 * 64, 256), 256, 0, stream>>>(x2o, cl2, enc, N2, 64);
  segmax_decode<<<cdiv(N3 * 64, 256), 256, 0, stream>>>(enc, x3in, x3bf, N3 * 64);

  // ---- layer 3 (out=128)
  cvt_swizzleB<<<cdiv(125 * 64 * 128, 256), 256, 0, stream>>>(W3, W3bf, 64, 128, 125);
  cvt_swizzleB<<<cdiv(64 * 128, 256), 256, 0, stream>>>(root3, r3bf, 64, 128, 1);
  gemm_sw_bf16<64><<<dim3(N3 / 16, 8, 125), 32, 0, stream>>>(x3bf, W3bf, 8,
                                                             XWb, 125 * 128, (long)128);
  gemm_sw_f32<64><<<dim3(N3 / 16, 8), 32, 0, stream>>>(x3bf, r3bf, ro3, 128, nullptr, 0);
  fill_u32<<<cdiv(N3 * 128, 256), 256, 0, stream>>>((unsigned*)agg3, 0u, N3 * 128);
  fill_u32<<<cdiv(N3, 256), 256, 0, stream>>>((unsigned*)deg3, 0u, N3);
  spline_conv_edges<<<cdiv(E3, 2), 256, 0, stream>>>(XWb, e3p, e3p + E3, ps3, agg3, deg3,
                                                     E3, 128, 125 * 128);
  finalize_g<<<cdiv(N3 * 128, 256), 256, 0, stream>>>(agg3, deg3, ro3, b3, x3o, N3, 128);

  // ---- pool3 -> (512,128) == (64,1024) row-major
  fill_u32<<<cdiv(512 * 128, 256), 256, 0, stream>>>(enc, ENC_NEGINF, 512 * 128);
  segmax_atomic<<<cdiv(N3 * 128, 256), 256, 0, stream>>>(x3o, cl3, enc, N3, 128);
  segmax_decode<<<cdiv(512 * 128, 256), 256, 0, stream>>>(enc, xfc, xfcbf, 512 * 128);

  // ---- FC1 (WMMA K=1024, fused bias+ELU) and FC2 + log_softmax
  cvt_swizzleB<<<cdiv(1024 * 256, 256), 256, 0, stream>>>(fc1w, fc1bf, 1024, 256, 1);
  gemm_sw_f32<1024><<<dim3(4, 16), 32, 0, stream>>>(xfcbf, fc1bf, h1, 256, fc1b, 1);
  fc2_logsoftmax<<<64, 32, 0, stream>>>(h1, fc2w, fc2b, out);
}
